// LocalAttention_53008486367939
// MI455X (gfx1250) — compile-verified, matrix operand order checked by reading, not run on hardware
//
#include <hip/hip_runtime.h>
#include <hip/hip_bf16.h>
#include <math.h>

#define BS 32
#define QLEN 128
#define NLEN 32
#define NUM_HEAD 8
#define DIM 32
#define D_DIST 64
#define F 256
#define HD 256
#define QB 4
#define ROWS (QB * NLEN)        /* 128 rows of `an` per workgroup */
#define LDA 264                  /* padded bf16 row stride for 256-col tiles */
#define LDW 264                  /* padded bf16 row stride for K=256 weights */
#define LDWD 72                  /* padded bf16 row stride for K=64 staging  */
#define SCALE_F 0.17677669529663687f /* 32^-0.5 */

typedef __attribute__((ext_vector_type(16))) __bf16 v16bf;
typedef __attribute__((ext_vector_type(8)))  __bf16 v8bf;
typedef __attribute__((ext_vector_type(8)))  float  v8f;

// Load one 16x32 bf16 A/B fragment from row-major LDS per the CDNA5 ISA layout:
// lanes 0-15 hold row (row0+lane), K = {kbase+0..7, kbase+16..23}
// lanes 16-31 hold row (row0+lane-16), K = {kbase+8..15, kbase+24..31}
__device__ __forceinline__ v16bf load_frag_bf16(const __bf16* base, int row0,
                                                int stride, int kbase, int lane) {
    int r  = row0 + (lane & 15);
    int ks = kbase + ((lane & 16) ? 8 : 0);
    const __bf16* p = base + r * stride + ks;
    v8bf lo = *(const v8bf*)(p);        // 16B -> ds_load_b128
    v8bf hi = *(const v8bf*)(p + 16);   // 16B -> ds_load_b128
    return __builtin_shufflevector(lo, hi, 0, 1, 2, 3, 4, 5, 6, 7,
                                           8, 9, 10, 11, 12, 13, 14, 15);
}

// Async global->LDS copy (CDNA5 GLOBAL_LOAD_ASYNC_TO_LDS_B32, ASYNCcnt-tracked).
// LDS destination = low 32 bits of the generic shared pointer.
__device__ __forceinline__ void async_copy_b32(void* lds_dst, const void* gsrc) {
    unsigned lds_off = (unsigned)(uintptr_t)lds_dst;
    asm volatile("global_load_async_to_lds_b32 %0, %1, off"
                 :: "v"(lds_off), "v"(gsrc) : "memory");
}
__device__ __forceinline__ void wait_async0() {
    asm volatile("s_wait_asynccnt 0x0" ::: "memory");
}

__global__ __launch_bounds__(256, 1)
void la_fused_kernel(const float* __restrict__ aq,     // (BS,QLEN,F)
                     const float* __restrict__ an_g,   // (BS,QLEN,NLEN,F)
                     const float* __restrict__ ldist,  // (BS,QLEN,NLEN,D_DIST)
                     const float* __restrict__ mask,   // (BS,QLEN,NLEN)
                     const float* __restrict__ Wd, const float* __restrict__ bd,
                     const float* __restrict__ Wq, const float* __restrict__ bq,
                     const float* __restrict__ Wk, const float* __restrict__ bk,
                     const float* __restrict__ Wv, const float* __restrict__ bv,
                     float* __restrict__ attn_out,     // (BS,NH,QLEN,NLEN,DIM)
                     float* __restrict__ ctx_out)      // (BS,QLEN,HD)
{
    __shared__ __align__(16) __bf16 sA[ROWS * LDA];  // an tile, bf16   (~66 KB)
    __shared__ __align__(16) __bf16 sWT[64 * LDW];   // weight chunk^T  (~33 KB)
    __shared__ __align__(16) __bf16 sK[ROWS * LDA];  // key tile        (~66 KB)
    __shared__ __align__(16) __bf16 sV[ROWS * LDA];  // value tile      (~66 KB)

    const int tid  = threadIdx.x;
    const int lane = tid & 31;
    const int w    = tid >> 5;                 // wave id, 0..7
    const int b    = blockIdx.x >> 5;          // batch
    const int q0   = (blockIdx.x & 31) * QB;   // first q of this block

    // ---------------- Phase 0: query rows (per-thread column) ----------------
    float* sAQ = (float*)sA;                   // temp alias: 4 x 256 f32
    for (int idx = tid; idx < QB * HD; idx += 256) {
        int qi = idx >> 8, c2 = idx & 255;
        async_copy_b32(&sAQ[idx], &aq[((size_t)(b * QLEN + q0 + qi)) * F + c2]);
    }
    wait_async0();
    __syncthreads();
    float qv[QB];                              // query[qi][col=tid], bias added
    {
        float a0 = 0.f, a1 = 0.f, a2 = 0.f, a3 = 0.f;
        #pragma unroll 8
        for (int k = 0; k < F; ++k) {
            float wq = Wq[(size_t)k * HD + tid];   // coalesced per wave
            a0 = fmaf(sAQ[0 * 256 + k], wq, a0);
            a1 = fmaf(sAQ[1 * 256 + k], wq, a1);
            a2 = fmaf(sAQ[2 * 256 + k], wq, a2);
            a3 = fmaf(sAQ[3 * 256 + k], wq, a3);
        }
        float bqv = bq[tid];
        qv[0] = a0 + bqv; qv[1] = a1 + bqv; qv[2] = a2 + bqv; qv[3] = a3 + bqv;
    }
    __syncthreads();

    // ---------------- Phase 1: ld = relu(LD@Wd+bd); an = neighbor*ld ---------
    __bf16* sLD = sK;                          // stage LD tile in sK region
    for (int idx = tid; idx < ROWS * D_DIST; idx += 256) {
        int row = idx >> 6, kk = idx & 63;
        int qi = row >> 5, n = row & 31;
        sLD[row * LDWD + kk] =
            (__bf16)ldist[(((size_t)(b * QLEN + q0 + qi)) * NLEN + n) * D_DIST + kk];
    }
    __syncthreads();
    // Hoist the per-wave LD A-fragments once; reused for all 16 output tiles.
    v16bf aLD[2];
    #pragma unroll
    for (int ks = 0; ks < 2; ++ks)
        aLD[ks] = load_frag_bf16(sLD, w * 16, LDWD, ks * 32, lane);

    for (int c = 0; c < 4; ++c) {              // 64-column output chunks
        __syncthreads();
        for (int idx = tid; idx < 64 * D_DIST; idx += 256) {
            int k = idx >> 6, nl = idx & 63;   // transpose Wd chunk -> sWT[n][k]
            sWT[nl * LDWD + k] = (__bf16)Wd[(size_t)k * HD + c * 64 + nl];
        }
        __syncthreads();
        #pragma unroll
        for (int nt = 0; nt < 4; ++nt) {       // 4 independent WMMA chains
            v8f acc = {};
            #pragma unroll
            for (int ks = 0; ks < 2; ++ks) {   // K = 64
                v16bf bfr = load_frag_bf16(sWT, nt * 16, LDWD, ks * 32, lane);
                acc = __builtin_amdgcn_wmma_f32_16x16x32_bf16(
                          false, aLD[ks], false, bfr, (short)0, acc, false, false);
            }
            int ncol = c * 64 + nt * 16 + (lane & 15);
            int mb   = w * 16 + ((lane & 16) ? 8 : 0);
            float bdv = bd[ncol];
            #pragma unroll
            for (int r = 0; r < 8; ++r) {
                int m = mb + r;
                int qi = m >> 5, n = m & 31;
                float x = fmaxf(acc[r] + bdv, 0.0f);
                float nb = an_g[(((size_t)(b * QLEN + q0 + qi)) * NLEN + n) * F + ncol];
                sA[m * LDA + ncol] = (__bf16)(x * nb);
            }
        }
    }

    // ---------------- Phase 2: key = an@Wk+bk ; value = an@Wv+bv -------------
    __syncthreads();
    // Hoist the per-wave `an` A-fragments once; invariant across both GEMMs,
    // all 4 column chunks and all 4 n-tiles (32 reuses per fragment).
    v16bf aAN[8];
    #pragma unroll
    for (int ks = 0; ks < 8; ++ks)
        aAN[ks] = load_frag_bf16(sA, w * 16, LDA, ks * 32, lane);

    for (int sel = 0; sel < 2; ++sel) {
        const float* W    = sel ? Wv : Wk;
        const float* bias = sel ? bv : bk;
        __bf16*      sOut = sel ? sV : sK;
        for (int c = 0; c < 4; ++c) {
            __syncthreads();
            __builtin_prefetch(W + (size_t)(((c + 1) & 3) * 64), 0, 1);
            for (int idx = tid; idx < 64 * F; idx += 256) {
                int k = idx >> 6, nl = idx & 63;  // transpose chunk -> sWT[n][k]
                sWT[nl * LDW + k] = (__bf16)W[(size_t)k * HD + c * 64 + nl];
            }
            __syncthreads();
            #pragma unroll
            for (int nt = 0; nt < 4; ++nt) {      // 4 independent WMMA chains
                v8f acc = {};
                #pragma unroll
                for (int ks = 0; ks < 8; ++ks) {  // K = 256
                    v16bf bfr = load_frag_bf16(sWT, nt * 16, LDW, ks * 32, lane);
                    acc = __builtin_amdgcn_wmma_f32_16x16x32_bf16(
                              false, aAN[ks], false, bfr, (short)0, acc, false, false);
                }
                int ncol = c * 64 + nt * 16 + (lane & 15);
                int mb   = w * 16 + ((lane & 16) ? 8 : 0);
                float bsv = bias[ncol];
                #pragma unroll
                for (int r = 0; r < 8; ++r)
                    sOut[(mb + r) * LDA + ncol] = (__bf16)(acc[r] + bsv);
            }
        }
    }
    __syncthreads();

    // ---------------- Phase 3: masked softmax over n + context ---------------
    float* sM = (float*)sWT;                   // alias: 128 mask values
    if (tid < ROWS) {
        int qi = tid >> 5, n = tid & 31;
        async_copy_b32(&sM[tid], &mask[((size_t)(b * QLEN + q0 + qi)) * NLEN + n]);
    }
    wait_async0();
    __syncthreads();

    const int h = tid >> 5, d = tid & 31;      // col = tid = h*32+d
    for (int j = 0; j < QB; ++j) {
        const int qi = j;
        const float qvv = qv[j];
        float e[NLEN];
        float mx = -3.4e38f;
        #pragma unroll
        for (int n = 0; n < NLEN; ++n) {
            float kx = (float)sK[(qi * NLEN + n) * LDA + tid];
            float en = sM[qi * NLEN + n] * qvv * kx * SCALE_F;
            e[n] = en;
            mx = fmaxf(mx, en);
        }
        float s = 0.f;
        #pragma unroll
        for (int n = 0; n < NLEN; ++n) { e[n] = __expf(e[n] - mx); s += e[n]; }
        float inv = 1.0f / s;
        float ctx = 0.f;
        size_t attn_base =
            ((((size_t)b * NUM_HEAD + h) * QLEN + (q0 + qi)) * NLEN) * DIM + d;
        #pragma unroll
        for (int n = 0; n < NLEN; ++n) {
            float a = e[n] * inv;
            attn_out[attn_base + (size_t)n * DIM] = a;   // 128B/lane-row, coalesced
            float vx = (float)sV[(qi * NLEN + n) * LDA + tid];
            ctx = fmaf(a, sM[qi * NLEN + n] * vx, ctx);
        }
        ctx_out[((size_t)(b * QLEN + q0 + qi)) * HD + tid] = ctx + qvv;
    }
}

extern "C" void kernel_launch(void* const* d_in, const int* in_sizes, int n_in,
                              void* d_out, int out_size, void* d_ws, size_t ws_size,
                              hipStream_t stream) {
    const float* aq    = (const float*)d_in[0];
    const float* an_g  = (const float*)d_in[1];
    const float* ldist = (const float*)d_in[2];
    const float* mask  = (const float*)d_in[3];
    const float* Wd    = (const float*)d_in[4];
    const float* bd    = (const float*)d_in[5];
    const float* Wq    = (const float*)d_in[6];
    const float* bq    = (const float*)d_in[7];
    const float* Wk    = (const float*)d_in[8];
    const float* bk    = (const float*)d_in[9];
    const float* Wv    = (const float*)d_in[10];
    const float* bv    = (const float*)d_in[11];

    float* attn = (float*)d_out;
    float* ctx  = attn + (size_t)BS * NUM_HEAD * QLEN * NLEN * DIM;

    dim3 grid(BS * (QLEN / QB));   // 1024 workgroups
    dim3 block(256);               // 8 wave32
    la_fused_kernel<<<grid, block, 0, stream>>>(aq, an_g, ldist, mask,
                                                Wd, bd, Wq, bq, Wk, bk, Wv, bv,
                                                attn, ctx);
}
